// RecSysModel_85169201480446
// MI455X (gfx1250) — matrix operand order repaired and checked by reading, not compiled, not run on hardware
//
#include <hip/hip_runtime.h>
#include <hip/hip_bf16.h>

typedef __attribute__((ext_vector_type(16))) __bf16 v16bf;
typedef __attribute__((ext_vector_type(8)))  float  v8f;
typedef unsigned short u16;
typedef unsigned int   u32;

#define ALPHA_C 0.5f

#if defined(__HIP_DEVICE_COMPILE__) && __has_builtin(__builtin_amdgcn_global_load_async_to_lds_b128)
#define HAS_ASYNC_LDS 1
#else
#define HAS_ASYNC_LDS 0
#endif

__device__ __forceinline__ void wait_async0() {
#if defined(__HIP_DEVICE_COMPILE__)
#if __has_builtin(__builtin_amdgcn_s_wait_asynccnt)
  __builtin_amdgcn_s_wait_asynccnt(0);
#else
  asm volatile("s_wait_asynccnt 0x0" ::: "memory");
#endif
#endif
}

// async 16-byte global -> LDS copy (ASYNCcnt-tracked); falls back to a
// register copy when the builtin is unavailable.
__device__ __forceinline__ void async_copy16(const u16* gsrc, u16* ldst) {
#if HAS_ASYNC_LDS
  typedef int v4i __attribute__((vector_size(16)));
  typedef __attribute__((address_space(1))) v4i* gp_t;
  typedef __attribute__((address_space(3))) v4i* lp_t;
  __builtin_amdgcn_global_load_async_to_lds_b128((gp_t)gsrc, (lp_t)ldst, 0, 0);
#else
  *(uint4*)ldst = *(const uint4*)gsrc;
#endif
}

// round-to-nearest-even fp32 -> bf16
__device__ __forceinline__ u16 f2bf(float f) {
  u32 u = __float_as_uint(f);
  u32 r = u + 0x7FFFu + ((u >> 16) & 1u);
  return (u16)(r >> 16);
}

// 16-element bf16 fragment from two 16-byte LDS chunks
__device__ __forceinline__ v16bf ld_frag(const u16* p0, const u16* p1) {
  union { v16bf v; uint4 q[2]; } u;
  u.q[0] = *(const uint4*)p0;
  u.q[1] = *(const uint4*)p1;
  return u.v;
}

// ---------------------------------------------------------------------------
// Stage 1 GEMMs (split-K, double-buffered): P1[s] += x@W1^T, P2[s] += x@W2^T
// over K chunk s. M=1280 (20 blocks x 64 rows), N=128. Epilogue deferred to
// encoder_reduce.
// ---------------------------------------------------------------------------
__global__ void __launch_bounds__(256)
encoder_gemm(const float* __restrict__ x, const float* __restrict__ W1,
             const float* __restrict__ W2, float* __restrict__ P1,
             float* __restrict__ P2, int KC)
{
  __shared__ __align__(16) u16 Ax[2][64 * 32];
  __shared__ __align__(16) u16 B1[2][128 * 32];
  __shared__ __align__(16) u16 B2[2][128 * 32];

  const int tid  = threadIdx.x;
  const int lane = tid & 31;
  const int w    = tid >> 5;
  const int m0   = blockIdx.x * 64;
  const int kbeg = blockIdx.y * KC;
  const int NIT  = KC >> 5;

  const int mloc = lane & 15;
  const int kb   = (lane >> 4) * 8;
  const int k0l  = (lane >> 4) * 16;
  const int ncol = w * 16 + (lane & 15);

  const int xm = tid >> 2, xks = (tid & 3) * 8;
  const int wn = tid >> 1, wks = (tid & 1) * 16;

  v8f z = {};
  v8f acc1[4] = {z, z, z, z};
  v8f acc2[4] = {z, z, z, z};

  auto stage = [&](int it, int buf) {
    const int k0 = kbeg + it * 32;
    const float* xp = x + (size_t)(m0 + xm) * 8192 + k0 + xks;
#pragma unroll
    for (int i = 0; i < 8; ++i) Ax[buf][xm * 32 + xks + i] = f2bf(xp[i]);
    const float* w1p = W1 + (size_t)wn * 8192 + k0 + wks;
    const float* w2p = W2 + (size_t)wn * 8192 + k0 + wks;
#pragma unroll
    for (int i = 0; i < 16; ++i) {
      B1[buf][wn * 32 + wks + i] = f2bf(w1p[i]);
      B2[buf][wn * 32 + wks + i] = f2bf(w2p[i]);
    }
  };

  stage(0, 0);
  __syncthreads();
  for (int it = 0; it < NIT; ++it) {
    const int buf = it & 1;
    if (it + 1 < NIT) stage(it + 1, buf ^ 1);

    v16bf fb1 = ld_frag(&B1[buf][ncol * 32 + k0l], &B1[buf][ncol * 32 + k0l + 8]);
    v16bf fb2 = ld_frag(&B2[buf][ncol * 32 + k0l], &B2[buf][ncol * 32 + k0l + 8]);
#pragma unroll
    for (int mt = 0; mt < 4; ++mt) {
      const int row = mt * 16 + mloc;
      v16bf fa = ld_frag(&Ax[buf][row * 32 + k0l], &Ax[buf][row * 32 + k0l + 8]);
      acc1[mt] = __builtin_amdgcn_wmma_f32_16x16x32_bf16(
          false, fa, false, fb1, (short)0, acc1[mt], false, false);
      acc2[mt] = __builtin_amdgcn_wmma_f32_16x16x32_bf16(
          false, fa, false, fb2, (short)0, acc2[mt], false, false);
    }
    __syncthreads();
  }

  float* P1s = P1 + (size_t)blockIdx.y * 163840;  // 1280*128
  float* P2s = P2 + (size_t)blockIdx.y * 163840;
  const int mrow = 8 * (lane >> 4);
#pragma unroll
  for (int mt = 0; mt < 4; ++mt) {
#pragma unroll
    for (int i = 0; i < 8; ++i) {
      const int m = m0 + mt * 16 + mrow + i;
      P1s[(size_t)m * 128 + ncol] = acc1[mt][i];
      P2s[(size_t)m * 128 + ncol] = acc2[mt][i];
    }
  }
}

// basket = max(relu(sum_s P1 + b1), relu(sum_s P2 + b2))
__global__ void __launch_bounds__(256)
encoder_reduce(const float* __restrict__ P1, const float* __restrict__ P2,
               const float* __restrict__ b1, const float* __restrict__ b2,
               float* __restrict__ basket, int S)
{
  const int idx = blockIdx.x * 256 + threadIdx.x;  // < 163840
  const int n = idx & 127;
  float s1 = 0.f, s2 = 0.f;
  for (int s = 0; s < S; ++s) {
    s1 += P1[(size_t)s * 163840 + idx];
    s2 += P2[(size_t)s * 163840 + idx];
  }
  float e1 = fmaxf(s1 + b1[n], 0.f);
  float e2 = fmaxf(s2 + b2[n], 0.f);
  basket[idx] = fmaxf(e1, e2);
}

// ---------------------------------------------------------------------------
// Stage 2: LSTM, one workgroup per batch element (independent recurrences).
// ---------------------------------------------------------------------------
__global__ void __launch_bounds__(512)
lstm_kernel(const float* __restrict__ basket, const int* __restrict__ seq_len,
            const float* __restrict__ W_ih, const float* __restrict__ W_hh,
            const float* __restrict__ b_ih, const float* __restrict__ b_hh,
            float* __restrict__ h_n)
{
  __shared__ float xt[128], hb[128], cb[128], gates[512];
  const int b = blockIdx.x;
  const int j = threadIdx.x;
  const int T = 20;

  if (j < 128) { hb[j] = 0.f; cb[j] = 0.f; }
  const int   sl   = seq_len[b];
  const float bias = b_ih[j] + b_hh[j];
  const float* wi = W_ih + (size_t)j * 128;
  const float* wh = W_hh + (size_t)j * 128;
  __syncthreads();

  for (int t = 0; t < T; ++t) {
    if (j < 128) xt[j] = basket[((size_t)b * T + t) * 128 + j];
    __syncthreads();
    float g = bias;
#pragma unroll 4
    for (int k = 0; k < 128; ++k) g += xt[k] * wi[k] + hb[k] * wh[k];
    gates[j] = g;
    __syncthreads();
    if (j < 128 && t < sl) {
      float ig = 1.f / (1.f + __expf(-gates[j]));
      float fg = 1.f / (1.f + __expf(-gates[128 + j]));
      float gg = tanhf(gates[256 + j]);
      float og = 1.f / (1.f + __expf(-gates[384 + j]));
      float cn = fg * cb[j] + ig * gg;
      float hn = og * tanhf(cn);
      cb[j] = cn; hb[j] = hn;
    }
    __syncthreads();
  }
  if (j < 128) h_n[(size_t)b * 128 + j] = hb[j];
}

// ---------------------------------------------------------------------------
// Stage 3a: p = sigmoid(h_n @ W_out^T), stored fp32 + bf16.
// ---------------------------------------------------------------------------
__global__ void __launch_bounds__(256)
proj_sigmoid(const float* __restrict__ h_n, const float* __restrict__ W_out,
             float* __restrict__ p_f32, u16* __restrict__ p_bf16)
{
  __shared__ float hr[128];
  const int m = blockIdx.y;
  const int n = blockIdx.x * 256 + threadIdx.x;
  if (threadIdx.x < 128) hr[threadIdx.x] = h_n[(size_t)m * 128 + threadIdx.x];
  __syncthreads();
  const float* wo = W_out + (size_t)n * 128;
  float acc = 0.f;
#pragma unroll 4
  for (int k = 0; k < 128; ++k) acc += hr[k] * wo[k];
  float p = 1.f / (1.f + __expf(-acc));
  p_f32[(size_t)m * 8192 + n]  = p;
  p_bf16[(size_t)m * 8192 + n] = f2bf(p);
}

// ---------------------------------------------------------------------------
// Stage 3b: Q[s] = p @ A over K chunk s. M=64, N=8192 (64 x S2 blocks).
// Double-buffered LDS; p tile staged via GLOBAL_LOAD_ASYNC_TO_LDS; A tile
// converted fp32->bf16 and transposed with packed b32 LDS writes. Epilogue
// deferred to final_reduce.
// ---------------------------------------------------------------------------
__global__ void __launch_bounds__(256)
final_gemm(const u16* __restrict__ p_bf16, const float* __restrict__ Amat,
           float* __restrict__ Q, int KC)
{
  __shared__ __align__(16) u16 Pt[2][64 * 32];
  __shared__ __align__(16) u16 Bt[2][128 * 40];  // [n][k], stride 40 halves

  const int tid  = threadIdx.x;
  const int lane = tid & 31;
  const int w    = tid >> 5;
  const int n0   = blockIdx.x * 128;
  const int kbeg = blockIdx.y * KC;
  const int NIT  = KC >> 5;

  const int mloc = lane & 15;
  const int kb   = (lane >> 4) * 8;
  const int k0l  = (lane >> 4) * 16;
  const int nloc = w * 16 + (lane & 15);

  const int pm  = tid >> 2,  pks = (tid & 3) * 8;   // p tile: 16B/thread
  const int kk2 = tid >> 4,  nb8 = (tid & 15) * 8;  // A tile: 2 k-rows x 8 n

  v8f z = {};
  v8f acc[4] = {z, z, z, z};

  auto stage = [&](int it, int buf) {
    const int k0 = kbeg + it * 32;
    async_copy16(p_bf16 + (size_t)pm * 8192 + k0 + pks,
                 &Pt[buf][pm * 32 + pks]);
    const float* a0 = Amat + (size_t)(k0 + 2 * kk2) * 8192 + n0 + nb8;
    const float* a1 = a0 + 8192;
#pragma unroll
    for (int i = 0; i < 8; ++i) {
      u32 v = (u32)f2bf(a0[i]) | ((u32)f2bf(a1[i]) << 16);
      *(u32*)&Bt[buf][(nb8 + i) * 40 + 2 * kk2] = v;
    }
    __builtin_prefetch(a0 + (size_t)32 * 8192, 0, 0);  // next K tile
  };

  stage(0, 0);
  wait_async0();
  __syncthreads();
  for (int it = 0; it < NIT; ++it) {
    const int buf = it & 1;
    if (it + 1 < NIT) stage(it + 1, buf ^ 1);

    v16bf fb = ld_frag(&Bt[buf][nloc * 40 + k0l], &Bt[buf][nloc * 40 + k0l + 8]);
#pragma unroll
    for (int mt = 0; mt < 4; ++mt) {
      const int row = mt * 16 + mloc;
      v16bf fa = ld_frag(&Pt[buf][row * 32 + k0l], &Pt[buf][row * 32 + k0l + 8]);
      acc[mt] = __builtin_amdgcn_wmma_f32_16x16x32_bf16(
          false, fa, false, fb, (short)0, acc[mt], false, false);
    }
    wait_async0();
    __syncthreads();
  }

  float* Qs = Q + (size_t)blockIdx.y * 524288;  // 64*8192
  const int n = n0 + nloc;
  const int mrow = 8 * (lane >> 4);
#pragma unroll
  for (int mt = 0; mt < 4; ++mt) {
#pragma unroll
    for (int i = 0; i < 8; ++i) {
      const int m = mt * 16 + mrow + i;
      Qs[(size_t)m * 8192 + n] = acc[mt][i];
    }
  }
}

// out = 0.5*p + 0.5*(p*relu(I_B) + relu(sum_s Q))
__global__ void __launch_bounds__(256)
final_reduce(const float* __restrict__ Q, const float* __restrict__ p_f32,
             const float* __restrict__ I_B, float* __restrict__ out, int S)
{
  const int idx = blockIdx.x * 256 + threadIdx.x;  // < 524288
  const int n = idx & 8191;
  float q = 0.f;
  for (int s = 0; s < S; ++s) q += Q[(size_t)s * 524288 + idx];
  float p  = p_f32[idx];
  float ib = fmaxf(I_B[n], 0.f);
  out[idx] = (1.f - ALPHA_C) * p + ALPHA_C * (p * ib + fmaxf(q, 0.f));
}

// ---------------------------------------------------------------------------
extern "C" void kernel_launch(void* const* d_in, const int* in_sizes, int n_in,
                              void* d_out, int out_size, void* d_ws, size_t ws_size,
                              hipStream_t stream)
{
  (void)in_sizes; (void)n_in; (void)out_size;

  const float* x     = (const float*)d_in[0];
  const int*   seq   = (const int*)  d_in[1];
  const float* Amat  = (const float*)d_in[2];
  const float* W1    = (const float*)d_in[3];
  const float* b1    = (const float*)d_in[4];
  const float* W2    = (const float*)d_in[5];
  const float* b2    = (const float*)d_in[6];
  const float* W_ih  = (const float*)d_in[7];
  const float* W_hh  = (const float*)d_in[8];
  const float* b_ih  = (const float*)d_in[9];
  const float* b_hh  = (const float*)d_in[10];
  const float* W_out = (const float*)d_in[11];
  const float* I_B   = (const float*)d_in[12];

  // Split-K factors chosen so the workspace fits (deterministic per ws_size).
  const size_t base = 655360ull + 32768ull + 2097152ull + 1048576ull;
  int S2 = 4;
  while (S2 > 1 && base + 2ull * 655360ull + (size_t)S2 * 2097152ull > ws_size) S2 >>= 1;
  int S1 = 8;
  while (S1 > 1 && base + (size_t)S1 * 2ull * 655360ull + (size_t)S2 * 2097152ull > ws_size) S1 >>= 1;

  char* ws = (char*)d_ws;
  size_t off = 0;
  float* P1     = (float*)(ws + off); off += (size_t)S1 * 655360ull;
  float* P2     = (float*)(ws + off); off += (size_t)S1 * 655360ull;
  float* Q      = (float*)(ws + off); off += (size_t)S2 * 2097152ull;
  float* basket = (float*)(ws + off); off += 655360ull;
  float* h_n    = (float*)(ws + off); off += 32768ull;
  float* p_f32  = (float*)(ws + off); off += 2097152ull;
  u16*   p_bf16 = (u16*)  (ws + off);

  encoder_gemm<<<dim3(20, S1), 256, 0, stream>>>(x, W1, W2, P1, P2, 8192 / S1);
  encoder_reduce<<<640, 256, 0, stream>>>(P1, P2, b1, b2, basket, S1);
  lstm_kernel<<<64, 512, 0, stream>>>(basket, seq, W_ih, W_hh, b_ih, b_hh, h_n);
  proj_sigmoid<<<dim3(32, 64), 256, 0, stream>>>(h_n, W_out, p_f32, p_bf16);
  final_gemm<<<dim3(64, S2), 256, 0, stream>>>(p_bf16, Amat, Q, 8192 / S2);
  final_reduce<<<2048, 256, 0, stream>>>(Q, p_f32, I_B, (float*)d_out, S2);
}
